// SubNetwork_13357348290880
// MI455X (gfx1250) — matrix operand order repaired and checked by reading, not compiled, not run on hardware
//
#include <hip/hip_runtime.h>
#include <math.h>

// ---------------------------------------------------------------------------
// CDNA5 (gfx1250) wave32 WMMA: D(16x16 f32) = A(16x32 f16) x B(32x16 f16) + C
// ---------------------------------------------------------------------------
typedef __attribute__((ext_vector_type(16))) _Float16 v16h;
typedef __attribute__((ext_vector_type(8)))  _Float16 v8h;
typedef __attribute__((ext_vector_type(8)))  float    v8f;

__device__ __forceinline__ v8f wmma_f32_f16(v16h a, v16h b, v8f c) {
    return __builtin_amdgcn_wmma_f32_16x16x32_f16(
        false, a, false, b, (short)0, c, false, false);
}

// A-fragment (16x32 f16) from f16 row-major W[*, ld].
// Lane layout: rows M = lane&15 (both halves); lanes<16 take K {kb=0..7,16..23},
// lanes>=16 take K {8..15,24..31}. Per lane: two contiguous 16B chunks.
__device__ __forceinline__ v16h load_a_frag_h(const _Float16* __restrict__ W,
                                              int ld, int row, int k0, int lane) {
    const int kb = (lane >> 4) * 8;
    const _Float16* wr = W + (size_t)row * ld + k0 + kb;
    v8h lo = *(const v8h*)(wr);        // K = k0+kb   .. +7
    v8h hi = *(const v8h*)(wr + 16);   // K = k0+16+kb.. +7
    return __builtin_shufflevector(lo, hi, 0, 1, 2, 3, 4, 5, 6, 7,
                                           8, 9, 10, 11, 12, 13, 14, 15);
}

// B-fragment (32x16 f16) from K-contiguous ("transposed") storage Xt[col][k],
// pitch ld. Lane holds column col, K = k0 + (lane>=16?8:0) + {0..7,16..23}:
// two 16-byte contiguous runs -> two b128 loads.
__device__ __forceinline__ v16h load_b_frag_h(const _Float16* __restrict__ Xt,
                                              int ld, int col, int k0, int lane) {
    const _Float16* p = Xt + (size_t)col * ld + k0 + (lane >> 4) * 8;
    v8h lo = *(const v8h*)(p);
    v8h hi = *(const v8h*)(p + 16);
    return __builtin_shufflevector(lo, hi, 0, 1, 2, 3, 4, 5, 6, 7,
                                           8, 9, 10, 11, 12, 13, 14, 15);
}

// ---------------------------------------------------------------------------
// f32 -> f16 weight conversion (run once per weight tensor)
// ---------------------------------------------------------------------------
__global__ void f32_to_f16(const float* __restrict__ s, _Float16* __restrict__ d, int n) {
    const int i = blockIdx.x * blockDim.x + threadIdx.x;
    if (i < n) d[i] = (_Float16)s[i];
}

// ---------------------------------------------------------------------------
// Generic conv1d-as-GEMM. Wh: f16 [O,I]. Xt: f16 transposed [B][Ncols][I].
// Outputs: optional f32 row-major Y[B][O][Ncols] (+residual) and/or
// f16 transposed Yt[B][Ncols][O] (feeds the next GEMM's B operand).
// One wave per 16x16 tile.
// ---------------------------------------------------------------------------
__global__ void wmma_gemm2(const _Float16* __restrict__ Wh, const _Float16* __restrict__ Xt,
                           const float* __restrict__ bias, const float* __restrict__ residual,
                           float* __restrict__ Yf, _Float16* __restrict__ Yt,
                           int O, int I, int Ncols, int relu) {
    const int lane = threadIdx.x & 31;
    const int n0 = blockIdx.x * 16;
    const int m0 = blockIdx.y * 16;
    const int b  = blockIdx.z;
    const _Float16* Xb = Xt + (size_t)b * Ncols * I;
    const int row = m0 + (lane & 15);
    const int col = n0 + (lane & 15);

    v8f acc = {};
    for (int k0 = 0; k0 < I; k0 += 32) {
        acc = wmma_f32_f16(load_a_frag_h(Wh, I, row, k0, lane),
                           load_b_frag_h(Xb, I, col, k0, lane), acc);
    }
    const int half = lane >> 4;
    float out[8];
#pragma unroll
    for (int r = 0; r < 8; ++r) {
        float v = acc[r] + bias[m0 + 8 * half + r];
        if (relu) v = fmaxf(v, 0.0f);
        out[r] = v;
    }
    if (Yf) {
        float* Yb = Yf + (size_t)b * O * Ncols;
        const float* Rb = residual ? residual + (size_t)b * O * Ncols : nullptr;
#pragma unroll
        for (int r = 0; r < 8; ++r) {
            const int oc = m0 + 8 * half + r;
            float v = out[r];
            if (Rb) v += Rb[(size_t)oc * Ncols + col];
            Yb[(size_t)oc * Ncols + col] = v;
        }
    }
    if (Yt) {
        v8h pk;
#pragma unroll
        for (int r = 0; r < 8; ++r) pk[r] = (_Float16)out[r];
        *(v8h*)(Yt + ((size_t)b * Ncols + col) * O + m0 + 8 * half) = pk;  // 16B store
    }
}

// ---------------------------------------------------------------------------
// mlp1 layer 0: 3 -> 64 + ReLU, writes f16 transposed [B][N][64]
// ---------------------------------------------------------------------------
__global__ void mlp1_l0t(const float* __restrict__ pts, const float* __restrict__ w,
                         const float* __restrict__ bias, _Float16* __restrict__ h1t, int N) {
    const int i = blockIdx.x * blockDim.x + threadIdx.x;
    const int total = 2 * N * 64;
    if (i >= total) return;
    const int o = i & 63;
    const int n = (i >> 6) % N;
    const int b = i / (N * 64);
    const float* pb = pts + (size_t)b * 3 * N + n;
    float v = w[o * 3 + 0] * pb[0] + w[o * 3 + 1] * pb[(size_t)N] +
              w[o * 3 + 2] * pb[(size_t)2 * N] + bias[o];
    h1t[i] = (_Float16)fmaxf(v, 0.0f);
}

// Global max over N per channel (input f16 transposed [B][N][128]).
__global__ void rowmax_t(const _Float16* __restrict__ f1t, float* __restrict__ g, int N) {
    const int b = blockIdx.x;
    const int ch = threadIdx.x;               // 128 threads
    const _Float16* p = f1t + (size_t)b * N * 128 + ch;
    float m = -1e30f;
    for (int n = 0; n < N; ++n) m = fmaxf(m, (float)p[(size_t)n * 128]);
    g[b * 128 + ch] = m;
}

// f2t[b][n][c] = c<128 ? f1t[b][n][c] : g[b][c-128]
__global__ void concat_t(const _Float16* __restrict__ f1t, const float* __restrict__ g,
                         _Float16* __restrict__ f2t, int N) {
    const int i = blockIdx.x * blockDim.x + threadIdx.x;
    const int total = 2 * N * 256;
    if (i >= total) return;
    const int c = i & 255;
    const int n = (i >> 8) % N;
    const int b = i / (N * 256);
    f2t[i] = (c < 128) ? f1t[((size_t)b * N + n) * 128 + c]
                       : (_Float16)g[b * 128 + (c - 128)];
}

// ---------------------------------------------------------------------------
// kNN (k=16). Points staged to LDS with CDNA5 async bulk copy
// (GLOBAL_LOAD_ASYNC_TO_LDS_B128, ASYNCcnt), then per-thread register top-16.
// ---------------------------------------------------------------------------
#define KNN 16
__global__ __launch_bounds__(256) void knn_kernel(const float* __restrict__ pts,
                                                  int* __restrict__ idx, int N) {
    const int b = blockIdx.y;
    __shared__ float spts[3 * 4096];          // same layout as pts[b][3][N]
    const float* px = pts + (size_t)b * 3 * N;
    {
        // 49152 bytes = 12 passes x 256 lanes x 16B, layout-preserving copy.
        const unsigned t = threadIdx.x;
        const unsigned ldsbase = (unsigned)(uintptr_t)(&spts[0]);  // low 32b = LDS offset
#pragma unroll
        for (int pass = 0; pass < 12; ++pass) {
            unsigned off = (pass * 256u + t) * 16u;
            unsigned lds = ldsbase + off;
            asm volatile("global_load_async_to_lds_b128 %0, %1, %2"
                         :: "v"(lds), "v"(off), "s"(px) : "memory");
        }
        asm volatile("s_wait_asynccnt 0x0" ::: "memory");
    }
    __syncthreads();
    const float* sx = spts, * sy = spts + 4096, * sz = spts + 8192;

    const int n = blockIdx.x * blockDim.x + threadIdx.x;
    const float qx = sx[n], qy = sy[n], qz = sz[n];
    float best[KNN];
    int bi[KNN];
#pragma unroll
    for (int s = 0; s < KNN; ++s) { best[s] = 1e30f; bi[s] = 0; }
    for (int m = 0; m < N; ++m) {
        const float dx = qx - sx[m], dy = qy - sy[m], dz = qz - sz[m];
        const float d = dx * dx + dy * dy + dz * dz;
        if (d < best[KNN - 1]) {
            float cd = d; int ci = m;
#pragma unroll
            for (int s = 0; s < KNN; ++s) {
                if (cd < best[s]) {
                    float td = best[s]; int ti = bi[s];
                    best[s] = cd; bi[s] = ci;
                    cd = td; ci = ti;
                }
            }
        }
    }
    int* op = idx + ((size_t)b * N + n) * KNN;
#pragma unroll
    for (int s = 0; s < KNN; ++s) op[s] = bi[s];
}

// ---------------------------------------------------------------------------
// Fused point-transformer attention. 256 threads (8 waves), 4 query points
// (64 WMMA columns = 4 pts x 16 neighbors). All LDS tiles are K-contiguous
// ("transposed") with padded pitches for conflict-free b128 access.
//
// Dynamic LDS carve (bytes):
//   sTt  [64][264] f16 : 33792  (qk_rel+pe, pitch 264 = 33x16B, conflict-free)
//   sPEt [64][264] f16 : 33792
//   sHt  [64][ 72] f16 :  9216
//   sUt  [64][ 72] f16 :  9216
//   sA   [256][65] f32 : 66560  (attention logits)
//   sIdx [64] int      :    256
//   sPR  [3][64] f32   :    768   -> total 153600 (< 320 KB/WGP)
// ---------------------------------------------------------------------------
__global__ __launch_bounds__(256) void fused_attn(
        const float* __restrict__ pts,
        const _Float16* __restrict__ qryt, const _Float16* __restrict__ keyt,
        const _Float16* __restrict__ valt, const int* __restrict__ knn,
        const float* __restrict__ pw0, const float* __restrict__ pb0,
        const float* __restrict__ pg0, const float* __restrict__ pbe0,
        const _Float16* __restrict__ pw1h, const float* __restrict__ pb1,
        const _Float16* __restrict__ aw0h, const float* __restrict__ ab0,
        const float* __restrict__ ag0, const float* __restrict__ abe0,
        const _Float16* __restrict__ aw1h, const float* __restrict__ ab1,
        _Float16* __restrict__ aggt, int N) {
    extern __shared__ char smem[];
    _Float16* sTt  = (_Float16*)(smem);
    _Float16* sPEt = (_Float16*)(smem + 33792);
    _Float16* sHt  = (_Float16*)(smem + 67584);
    _Float16* sUt  = (_Float16*)(smem + 76800);
    float*    sA   = (float*)   (smem + 86016);
    int*      sIdx = (int*)     (smem + 152576);
    float*    sPR  = (float*)   (smem + 152832);
    const int LDT = 264, LDH = 72, LDA = 65;

    const int b    = blockIdx.y;
    const int n0   = blockIdx.x * 4;
    const int tid  = threadIdx.x;
    const int lane = tid & 31;
    const int wave = tid >> 5;
    const int col_l = lane & 15, half = lane >> 4;
    const float inv = rsqrtf(1.0f + 1e-5f);     // BN running var = 1

    if (tid < 64)
        sIdx[tid] = knn[((size_t)b * N + n0 + (tid >> 4)) * KNN + (tid & 15)];
    __syncthreads();
    if (tid < 192) {
        const int d = tid / 64, col = tid % 64;
        const int n = n0 + (col >> 4), j = sIdx[col];
        const float* pd = pts + ((size_t)b * 3 + d) * N;
        sPR[d * 64 + col] = pd[n] - pd[j];
    }
    __syncthreads();
    // h = relu(bn(pw0 * pos_rel + pb0)), stored K-contiguous [col][c]
    for (int e = tid; e < 64 * 64; e += 256) {
        const int col = e >> 6, c = e & 63;
        float v = pw0[c * 3 + 0] * sPR[col] + pw0[c * 3 + 1] * sPR[64 + col] +
                  pw0[c * 3 + 2] * sPR[128 + col] + pb0[c];
        v = v * (pg0[c] * inv) + pbe0[c];
        sHt[col * LDH + c] = (_Float16)fmaxf(v, 0.0f);
    }
    __syncthreads();

    // pe = pw1*h + pb1 ; t = (q - key_gather) + pe. A-frags reused over 4 tiles.
#pragma unroll
    for (int i = 0; i < 2; ++i) {
        const int m0 = (wave * 2 + i) * 16;
        const int row = m0 + (lane & 15);
        const v16h a0 = load_a_frag_h(pw1h, 64, row, 0, lane);
        const v16h a1 = load_a_frag_h(pw1h, 64, row, 32, lane);
#pragma unroll
        for (int nt = 0; nt < 4; ++nt) {
            const int col = nt * 16 + col_l;
            v8f acc = {};
            acc = wmma_f32_f16(a0, load_b_frag_h(sHt, LDH, col, 0, lane), acc);
            acc = wmma_f32_f16(a1, load_b_frag_h(sHt, LDH, col, 32, lane), acc);
            const int n = n0 + (col >> 4), j = sIdx[col];
            // gathered q/key: channels contiguous at a point -> single 16B loads
            const v8h q8 = *(const v8h*)(qryt + ((size_t)b * N + n) * 256 + m0 + 8 * half);
            const v8h k8 = *(const v8h*)(keyt + ((size_t)b * N + j) * 256 + m0 + 8 * half);
            v8h pe8, t8;
#pragma unroll
            for (int r = 0; r < 8; ++r) {
                const float pe = acc[r] + pb1[m0 + 8 * half + r];
                const float qk = (float)q8[r] - (float)k8[r];
                pe8[r] = (_Float16)pe;
                t8[r]  = (_Float16)(qk + pe);
            }
            *(v8h*)(sPEt + col * LDT + m0 + 8 * half) = pe8;
            *(v8h*)(sTt  + col * LDT + m0 + 8 * half) = t8;
        }
    }
    __syncthreads();

    // a = aw1 * relu(bn(aw0 * t + ab0)) + ab1, 64-row hidden chunks
    v8f aacc[2][4] = {};
    for (int kb = 0; kb < 16; ++kb) {
        {   // u-chunk: wave -> (mt = wave>>1, two nt tiles); A-frags loaded once
            const int mt = wave >> 1;
            const int ntb = (wave & 1) * 2;
            const int hr0 = kb * 64 + mt * 16;
            const int row = hr0 + (lane & 15);
            __builtin_prefetch(aw0h + (size_t)(row + 64) * 256, 0, 1);
            v16h af[8];
#pragma unroll
            for (int ks = 0; ks < 8; ++ks)
                af[ks] = load_a_frag_h(aw0h, 256, row, ks * 32, lane);
#pragma unroll
            for (int t = 0; t < 2; ++t) {
                const int col = (ntb + t) * 16 + col_l;
                v8f acc = {};
#pragma unroll
                for (int ks = 0; ks < 8; ++ks)
                    acc = wmma_f32_f16(af[ks], load_b_frag_h(sTt, LDT, col, ks * 32, lane), acc);
                v8h u8;
#pragma unroll
                for (int r = 0; r < 8; ++r) {
                    const int h = hr0 + 8 * half + r;
                    const float u = (acc[r] + ab0[h]) * (ag0[h] * inv) + abe0[h];
                    u8[r] = (_Float16)fmaxf(u, 0.0f);
                }
                *(v8h*)(sUt + col * LDH + mt * 16 + 8 * half) = u8;
            }
        }
        __syncthreads();
#pragma unroll
        for (int i = 0; i < 2; ++i) {
            const int m0 = (wave * 2 + i) * 16;
            const int row = m0 + (lane & 15);
            const v16h a0 = load_a_frag_h(aw1h, 1024, row, kb * 64, lane);
            const v16h a1 = load_a_frag_h(aw1h, 1024, row, kb * 64 + 32, lane);
#pragma unroll
            for (int nt = 0; nt < 4; ++nt) {
                const int col = nt * 16 + col_l;
                v8f acc = aacc[i][nt];
                acc = wmma_f32_f16(a0, load_b_frag_h(sUt, LDH, col, 0, lane), acc);
                acc = wmma_f32_f16(a1, load_b_frag_h(sUt, LDH, col, 32, lane), acc);
                aacc[i][nt] = acc;
            }
        }
        __syncthreads();
    }

    // spill logits (+bias) for per-point softmax
#pragma unroll
    for (int i = 0; i < 2; ++i) {
        const int m0 = (wave * 2 + i) * 16;
#pragma unroll
        for (int nt = 0; nt < 4; ++nt) {
            const int col = nt * 16 + col_l;
#pragma unroll
            for (int r = 0; r < 8; ++r)
                sA[(m0 + 8 * half + r) * LDA + col] = aacc[i][nt][r] + ab1[m0 + 8 * half + r];
        }
    }
    __syncthreads();

    // softmax over k (per channel, per point) + aggregate with v_t = val + pe
    for (int e = tid; e < 256 * 4; e += 256) {
        const int ch = e >> 2, p = e & 3;
        const int n = n0 + p;
        const float* arow = sA + ch * LDA + p * 16;
        float m = arow[0];
#pragma unroll
        for (int k = 1; k < 16; ++k) m = fmaxf(m, arow[k]);
        float s = 0.0f, dot = 0.0f;
        const float vv = (float)valt[((size_t)b * N + n) * 256 + ch];
#pragma unroll
        for (int k = 0; k < 16; ++k) {
            const float ex = __expf(arow[k] - m);
            s += ex;
            dot += ex * (vv + (float)sPEt[(p * 16 + k) * LDT + ch]);
        }
        aggt[((size_t)b * N + n) * 256 + ch] = (_Float16)(dot / s);
    }
}

// ---------------------------------------------------------------------------
// Host launcher
// ---------------------------------------------------------------------------
extern "C" void kernel_launch(void* const* d_in, const int* in_sizes, int n_in,
                              void* d_out, int out_size, void* d_ws, size_t ws_size,
                              hipStream_t stream) {
    (void)in_sizes; (void)n_in; (void)out_size; (void)ws_size;
    const int B = 2, N = 4096;

    const float* points  = (const float*)d_in[0];
    const float* mlp1_w0 = (const float*)d_in[1];
    const float* mlp1_b0 = (const float*)d_in[2];
    const float* mlp1_w1 = (const float*)d_in[3];
    const float* mlp1_b1 = (const float*)d_in[4];
    const float* mlp2_w0 = (const float*)d_in[5];
    const float* mlp2_b0 = (const float*)d_in[6];
    const float* mlp2_w1 = (const float*)d_in[7];
    const float* mlp2_b1 = (const float*)d_in[8];
    const float* ls_w    = (const float*)d_in[9];
    const float* ls_b    = (const float*)d_in[10];
    const float* k_w     = (const float*)d_in[11];
    const float* k_b     = (const float*)d_in[12];
    const float* q_w     = (const float*)d_in[13];
    const float* q_b     = (const float*)d_in[14];
    const float* v_w     = (const float*)d_in[15];
    const float* v_b     = (const float*)d_in[16];
    const float* pos_w0  = (const float*)d_in[17];
    const float* pos_b0  = (const float*)d_in[18];
    const float* pos_g0  = (const float*)d_in[19];
    const float* pos_be0 = (const float*)d_in[20];
    const float* pos_w1  = (const float*)d_in[21];
    const float* pos_b1  = (const float*)d_in[22];
    const float* att_w0  = (const float*)d_in[23];
    const float* att_b0  = (const float*)d_in[24];
    const float* att_g0  = (const float*)d_in[25];
    const float* att_be0 = (const float*)d_in[26];
    const float* att_w1  = (const float*)d_in[27];
    const float* att_b1  = (const float*)d_in[28];
    const float* le_w    = (const float*)d_in[29];
    const float* le_b    = (const float*)d_in[30];

    char* base = (char*)d_ws;
    size_t off = 0;
    auto carve = [&](size_t bytes) {
        off = (off + 31) & ~(size_t)31;
        void* p = base + off;
        off += bytes;
        return p;
    };
    typedef _Float16 h16;
    // f16 weight copies
    h16* wh_m1  = (h16*)carve((size_t)128 * 64 * 2);
    h16* wh_m2a = (h16*)carve((size_t)256 * 256 * 2);
    h16* wh_m2b = (h16*)carve((size_t)128 * 256 * 2);
    h16* wh_ls  = (h16*)carve((size_t)256 * 128 * 2);
    h16* wh_k   = (h16*)carve((size_t)256 * 256 * 2);
    h16* wh_q   = (h16*)carve((size_t)256 * 256 * 2);
    h16* wh_v   = (h16*)carve((size_t)256 * 256 * 2);
    h16* wh_p1  = (h16*)carve((size_t)256 * 64 * 2);
    h16* wh_a0  = (h16*)carve((size_t)1024 * 256 * 2);
    h16* wh_a1  = (h16*)carve((size_t)256 * 1024 * 2);
    h16* wh_le  = (h16*)carve((size_t)128 * 256 * 2);
    // f16 transposed activations [B][N][C]
    h16* h1t  = (h16*)carve((size_t)B * N * 64 * 2);
    h16* f1t  = (h16*)carve((size_t)B * N * 128 * 2);
    h16* f2t  = (h16*)carve((size_t)B * N * 256 * 2);
    h16* f2at = (h16*)carve((size_t)B * N * 256 * 2);
    h16* f3t  = (h16*)carve((size_t)B * N * 128 * 2);
    h16* xt   = (h16*)carve((size_t)B * N * 256 * 2);
    h16* kt   = (h16*)carve((size_t)B * N * 256 * 2);
    h16* qt   = (h16*)carve((size_t)B * N * 256 * 2);
    h16* vt   = (h16*)carve((size_t)B * N * 256 * 2);
    h16* aggt = (h16*)carve((size_t)B * N * 256 * 2);
    // f32 / int scratch
    float* f3  = (float*)carve((size_t)B * 128 * N * 4);   // identity, row-major
    float* g   = (float*)carve((size_t)B * 128 * 4);
    int* idxb  = (int*)carve((size_t)B * N * KNN * 4);
    float* yout = (float*)d_out;

    // weight conversions
    struct { const float* s; h16* d; int n; } cv[11] = {
        {mlp1_w1, wh_m1, 128 * 64},  {mlp2_w0, wh_m2a, 256 * 256},
        {mlp2_w1, wh_m2b, 128 * 256},{ls_w, wh_ls, 256 * 128},
        {k_w, wh_k, 256 * 256},      {q_w, wh_q, 256 * 256},
        {v_w, wh_v, 256 * 256},      {pos_w1, wh_p1, 256 * 64},
        {att_w0, wh_a0, 1024 * 256}, {att_w1, wh_a1, 256 * 1024},
        {le_w, wh_le, 128 * 256}};
    for (int i = 0; i < 11; ++i)
        f32_to_f16<<<(cv[i].n + 255) / 256, 256, 0, stream>>>(cv[i].s, cv[i].d, cv[i].n);

    // mlp_1
    mlp1_l0t<<<(B * N * 64 + 255) / 256, 256, 0, stream>>>(points, mlp1_w0, mlp1_b0, h1t, N);
    wmma_gemm2<<<dim3(N / 16, 8, B), 32, 0, stream>>>(
        wh_m1, h1t, mlp1_b1, nullptr, nullptr, f1t, 128, 64, N, 0);
    // global feature + concat
    rowmax_t<<<B, 128, 0, stream>>>(f1t, g, N);
    concat_t<<<(B * N * 256 + 255) / 256, 256, 0, stream>>>(f1t, g, f2t, N);
    // mlp_2 (l1 emits both f32 identity and f16-transposed)
    wmma_gemm2<<<dim3(N / 16, 16, B), 32, 0, stream>>>(
        wh_m2a, f2t, mlp2_b0, nullptr, nullptr, f2at, 256, 256, N, 1);
    wmma_gemm2<<<dim3(N / 16, 8, B), 32, 0, stream>>>(
        wh_m2b, f2at, mlp2_b1, nullptr, f3, f3t, 128, 256, N, 0);
    // linear_start
    wmma_gemm2<<<dim3(N / 16, 16, B), 32, 0, stream>>>(
        wh_ls, f3t, ls_b, nullptr, nullptr, xt, 256, 128, N, 0);
    // kNN (async-to-LDS staging)
    knn_kernel<<<dim3(N / 256, B), 256, 0, stream>>>(points, idxb, N);
    // k / q / v projections
    wmma_gemm2<<<dim3(N / 16, 16, B), 32, 0, stream>>>(
        wh_k, xt, k_b, nullptr, nullptr, kt, 256, 256, N, 0);
    wmma_gemm2<<<dim3(N / 16, 16, B), 32, 0, stream>>>(
        wh_q, xt, q_b, nullptr, nullptr, qt, 256, 256, N, 0);
    wmma_gemm2<<<dim3(N / 16, 16, B), 32, 0, stream>>>(
        wh_v, xt, v_b, nullptr, nullptr, vt, 256, 256, N, 0);
    // fused pos-enc + attention MLP + softmax + aggregation
    fused_attn<<<dim3(N / 4, B), 256, 153600, stream>>>(
        points, qt, kt, vt, idxb,
        pos_w0, pos_b0, pos_g0, pos_be0, wh_p1, pos_b1,
        wh_a0, att_b0, att_g0, att_be0, wh_a1, att_b1,
        aggt, N);
    // linear_end + residual
    wmma_gemm2<<<dim3(N / 16, 8, B), 32, 0, stream>>>(
        wh_le, aggt, le_b, f3, yout, nullptr, 128, 256, N, 0);
}
